// TripletLossAlignedReID_9131100471769
// MI455X (gfx1250) — matrix operand order repaired and checked by reading, not compiled, not run on hardware
//
#include <hip/hip_runtime.h>
#include <hip/hip_bf16.h>
#include <math.h>

#define N_      4096
#define D_      2048
#define C_      128
#define P_      8
#define MARGIN_ 0.3f
#define EPS_    1e-12f

// Gram/mining blocking
#define BM_     64        // rows per workgroup
#define BN_     128       // staged column block
#define KC_     128       // staged K chunk
#define SPLITS_ 4         // column splits across workgroups
#define COLSPAN_ (N_ / SPLITS_)          // 1024 columns per WG
#define NCB_    (COLSPAN_ / BN_)         // 8 column blocks per WG
#define NKB_    (D_ / KC_)               // 16 K chunks
#define NSTAGES_ (NCB_ * NKB_)           // 128 pipeline stages
#define A_ELEMS_ (BM_ * KC_)             // 8192 bf16  (16 KB)
#define B_ELEMS_ (BN_ * KC_)             // 16384 bf16 (32 KB)
#define STAGE_ELEMS_ (A_ELEMS_ + B_ELEMS_)  // 24576 bf16 (48 KB)

typedef __attribute__((ext_vector_type(16))) __bf16 v16bf;
typedef __attribute__((ext_vector_type(8)))  __bf16 v8bf;
typedef __attribute__((ext_vector_type(8)))  float  v8f;

// ---------------------------------------------------------------------------
// Kernel 1: L2-normalize each row of inputs, emit bf16 copy for WMMA.
// ---------------------------------------------------------------------------
__global__ void k_normalize(const float* __restrict__ in, __bf16* __restrict__ outb) {
  const int row = blockIdx.x;
  const float* r = in + (size_t)row * D_;
  float ss = 0.f;
  for (int c = threadIdx.x; c < D_; c += blockDim.x) { float x = r[c]; ss += x * x; }
  __shared__ float red[256];
  red[threadIdx.x] = ss;
  __syncthreads();
  for (int s = 128; s > 0; s >>= 1) {
    if (threadIdx.x < s) red[threadIdx.x] += red[threadIdx.x + s];
    __syncthreads();
  }
  const float inv = 1.0f / (sqrtf(red[0]) + EPS_);
  __bf16* o = outb + (size_t)row * D_;
  for (int c = threadIdx.x; c < D_; c += blockDim.x) o[c] = (__bf16)(r[c] * inv);
}

// ---------------------------------------------------------------------------
// Async staging of one (A 64xKC, B 128xKC) chunk into an LDS buffer using
// CDNA5 global_load_async_to_lds_b128 (ASYNCcnt-tracked, no VGPR bounce).
// ---------------------------------------------------------------------------
__device__ __forceinline__ void stage_chunk(const __bf16* __restrict__ featsb,
                                            uint32_t ldsBase, int buf,
                                            int i0, int j0, int kb, int tid) {
  const uint32_t aBase = ldsBase + (uint32_t)(buf * (STAGE_ELEMS_ * 2));
  const uint32_t bBase = aBase + (uint32_t)(A_ELEMS_ * 2);
#pragma unroll
  for (int q = 0; q < 4; ++q) {           // A: 1024 x 16B chunks
    const int c = tid + q * 256;
    const int row = c >> 4;               // 16 chunks per 256B row
    const int ko  = (c & 15) * 8;
    const void* g = featsb + (size_t)(i0 + row) * D_ + kb + ko;
    asm volatile("global_load_async_to_lds_b128 %0, %1, off"
                 :: "v"(aBase + (uint32_t)(c * 16)), "v"(g) : "memory");
  }
#pragma unroll
  for (int q = 0; q < 8; ++q) {           // B: 2048 x 16B chunks
    const int c = tid + q * 256;
    const int row = c >> 4;
    const int ko  = (c & 15) * 8;
    const void* g = featsb + (size_t)(j0 + row) * D_ + kb + ko;
    asm volatile("global_load_async_to_lds_b128 %0, %1, off"
                 :: "v"(bBase + (uint32_t)(c * 16)), "v"(g) : "memory");
  }
}

// A fragment for v_wmma_..._bf16: two 16B LDS chunks at k+{0,8} (+16 hi lanes).
__device__ __forceinline__ v16bf load_afrag(const __bf16* aRow, int kc, int aOfs) {
  v8bf lo = *(const v8bf*)(aRow + kc + aOfs);
  v8bf hi = *(const v8bf*)(aRow + kc + aOfs + 8);
  return __builtin_shufflevector(lo, hi,
      0, 1, 2, 3, 4, 5, 6, 7, 8, 9, 10, 11, 12, 13, 14, 15);
}

// ---------------------------------------------------------------------------
// Kernel 2: fused bf16 WMMA Gram + hard mining, cooperative LDS staging.
// Grid: (N/BM) x SPLITS workgroups, 256 threads (8 waves = 4 row-subtiles x
// 2 column halves). Double-buffered async pipeline over 128 stages, with
// one-kc-step fragment lookahead so LDS loads overlap WMMA execution.
// dist(i,j) = sqrt(clip(2 - 2*gram, eps)) since rows are unit-normalized.
// ---------------------------------------------------------------------------
__global__ void __launch_bounds__(256) k_gram_mine(
    const __bf16* __restrict__ featsb, const int* __restrict__ targets,
    float* __restrict__ pAP, float* __restrict__ pAN,
    int* __restrict__ pPI, int* __restrict__ pNI) {
  __shared__ __align__(32) __bf16 smem[2 * STAGE_ELEMS_];   // 96 KB
  const int tid    = threadIdx.x;
  const int lane   = tid & 31;
  const int wave   = tid >> 5;
  const int lane15 = lane & 15;
  const int laneHi = lane >> 4;
  const int wr     = wave >> 1;            // row sub-tile 0..3
  const int wc     = wave & 1;             // column half 0..1
  const int by     = blockIdx.x >> 2;      // row block
  const int bx     = blockIdx.x & 3;       // column split
  const int i0     = by * BM_;
  const int colbase = bx * COLSPAN_;
  const uint32_t ldsBase = (uint32_t)(uintptr_t)(void*)smem;

  int trow[8];
#pragma unroll
  for (int v = 0; v < 8; ++v) trow[v] = targets[i0 + wr * 16 + v + (laneHi ? 8 : 0)];

  float bestp[8], bestn[8];
  int bpi[8], bni[8];
#pragma unroll
  for (int v = 0; v < 8; ++v) { bestp[v] = -3.0e38f; bestn[v] = 3.0e38f; bpi[v] = 0; bni[v] = 0; }

  const int aOfs = laneHi ? 16 : 0;   // A frag hi-lane K offset
  const int bOfs = laneHi ? 16 : 0;   // B frag: lanes 0-15 K=0..15, 16-31 K=16..31
  const v8f vzero = {0.f, 0.f, 0.f, 0.f, 0.f, 0.f, 0.f, 0.f};
  v8f acc[4];
#pragma unroll
  for (int t = 0; t < 4; ++t) acc[t] = vzero;

  // Prime the pipeline: stage 0 -> buffer 0.
  stage_chunk(featsb, ldsBase, 0, i0, colbase, 0, tid);

  for (int st = 0; st < NSTAGES_; ++st) {
    const int buf = st & 1;
    asm volatile("s_wait_asynccnt 0x0" ::: "memory");  // my async stores landed
    __syncthreads();                                   // everyone's landed; prev buf free
    if (st + 1 < NSTAGES_) {
      const int nst = st + 1;
      stage_chunk(featsb, ldsBase, buf ^ 1, i0,
                  colbase + (nst >> 4) * BN_, (nst & 15) * KC_, tid);
    }
    // Compute from staged buffer with one-step fragment lookahead:
    // issue next kc's LDS loads before current kc's 4 WMMAs.
    const __bf16* sA = smem + buf * STAGE_ELEMS_;
    const __bf16* sB = sA + A_ELEMS_;
    const __bf16* aRow  = sA + (wr * 16 + lane15) * KC_;
    const __bf16* bRow0 = sB + (wc * 64 + lane15) * KC_ + bOfs;

    v16bf aN  = load_afrag(aRow, 0, aOfs);
    v16bf bN0 = *(const v16bf*)(bRow0 + 0 * 16 * KC_);
    v16bf bN1 = *(const v16bf*)(bRow0 + 1 * 16 * KC_);
    v16bf bN2 = *(const v16bf*)(bRow0 + 2 * 16 * KC_);
    v16bf bN3 = *(const v16bf*)(bRow0 + 3 * 16 * KC_);
#pragma unroll
    for (int kc = 0; kc < KC_; kc += 32) {
      v16bf aC = aN, bC0 = bN0, bC1 = bN1, bC2 = bN2, bC3 = bN3;
      if (kc + 32 < KC_) {
        aN  = load_afrag(aRow, kc + 32, aOfs);
        bN0 = *(const v16bf*)(bRow0 + kc + 32 + 0 * 16 * KC_);
        bN1 = *(const v16bf*)(bRow0 + kc + 32 + 1 * 16 * KC_);
        bN2 = *(const v16bf*)(bRow0 + kc + 32 + 2 * 16 * KC_);
        bN3 = *(const v16bf*)(bRow0 + kc + 32 + 3 * 16 * KC_);
      }
      acc[0] = __builtin_amdgcn_wmma_f32_16x16x32_bf16(
          false, aC, false, bC0, (short)0, acc[0], false, false);
      acc[1] = __builtin_amdgcn_wmma_f32_16x16x32_bf16(
          false, aC, false, bC1, (short)0, acc[1], false, false);
      acc[2] = __builtin_amdgcn_wmma_f32_16x16x32_bf16(
          false, aC, false, bC2, (short)0, acc[2], false, false);
      acc[3] = __builtin_amdgcn_wmma_f32_16x16x32_bf16(
          false, aC, false, bC3, (short)0, acc[3], false, false);
    }
    // Column block finished (every 16 K-chunks): mine and reset accumulators.
    if ((st & 15) == 15) {
      const int j0 = colbase + (st >> 4) * BN_;
#pragma unroll
      for (int t = 0; t < 4; ++t) {
        const int coln = j0 + wc * 64 + t * 16 + lane15;
        const int tcol = targets[coln];
#pragma unroll
        for (int v = 0; v < 8; ++v) {
          float g = acc[t][v];
          float d = sqrtf(fmaxf(2.0f - 2.0f * g, EPS_));
          bool pos = (tcol == trow[v]);
          float pv = pos ? d : -3.0e38f;
          float nv = pos ? 3.0e38f : d;
          if (pv > bestp[v]) { bestp[v] = pv; bpi[v] = coln; }
          if (nv < bestn[v]) { bestn[v] = nv; bni[v] = coln; }
        }
        acc[t] = vzero;
      }
    }
  }

  // Butterfly-reduce (value,index) across the 16 lanes holding each row.
#pragma unroll
  for (int v = 0; v < 8; ++v) {
#pragma unroll
    for (int m = 8; m >= 1; m >>= 1) {
      float ov = __shfl_xor(bestp[v], m, 32);
      int   oi = __shfl_xor(bpi[v],   m, 32);
      if (ov > bestp[v]) { bestp[v] = ov; bpi[v] = oi; }
      float on  = __shfl_xor(bestn[v], m, 32);
      int   oni = __shfl_xor(bni[v],   m, 32);
      if (on < bestn[v]) { bestn[v] = on; bni[v] = oni; }
    }
  }

  // Combine the two column-half waves per row via LDS overlay on dead smem.
  __syncthreads();
  float* redAP = (float*)smem;        // [2][64]
  float* redAN = redAP + 128;
  int*   redPI = (int*)(redAN + 128);
  int*   redNI = redPI + 128;
  if (lane15 == 0) {
#pragma unroll
    for (int v = 0; v < 8; ++v) {
      const int r = wr * 16 + v + (laneHi ? 8 : 0);
      redAP[wc * 64 + r] = bestp[v];
      redAN[wc * 64 + r] = bestn[v];
      redPI[wc * 64 + r] = bpi[v];
      redNI[wc * 64 + r] = bni[v];
    }
  }
  __syncthreads();
  if (tid < BM_) {
    float bp = -3.0e38f, bn = 3.0e38f;
    int pi = 0, ni = 0;
#pragma unroll
    for (int h = 0; h < 2; ++h) {
      float vP = redAP[h * 64 + tid];
      if (vP > bp) { bp = vP; pi = redPI[h * 64 + tid]; }
      float vN = redAN[h * 64 + tid];
      if (vN < bn) { bn = vN; ni = redNI[h * 64 + tid]; }
    }
    const int r = i0 + tid;
    pAP[bx * N_ + r] = bp;
    pAN[bx * N_ + r] = bn;
    pPI[bx * N_ + r] = pi;
    pNI[bx * N_ + r] = ni;
  }
}

// ---------------------------------------------------------------------------
// Kernel 2b: merge the SPLITS_ per-split mining partials.
// ---------------------------------------------------------------------------
__global__ void k_mine_combine(
    const float* __restrict__ pAP, const float* __restrict__ pAN,
    const int* __restrict__ pPI, const int* __restrict__ pNI,
    float* __restrict__ dist_ap, float* __restrict__ dist_an,
    int* __restrict__ p_inds, int* __restrict__ n_inds) {
  const int r = blockIdx.x * blockDim.x + threadIdx.x;
  float bp = -3.0e38f, bn = 3.0e38f;
  int pi = 0, ni = 0;
#pragma unroll
  for (int s = 0; s < SPLITS_; ++s) {
    float vP = pAP[s * N_ + r];
    if (vP > bp) { bp = vP; pi = pPI[s * N_ + r]; }
    float vN = pAN[s * N_ + r];
    if (vN < bn) { bn = vN; ni = pNI[s * N_ + r]; }
  }
  dist_ap[r] = bp;
  dist_an[r] = bn;
  p_inds[r] = pi;
  n_inds[r] = ni;
}

// ---------------------------------------------------------------------------
// Kernel 3: local aligned distance (one wave per (sample, branch); 32 lanes
// cover the 64 part-pair dot products; lane 0 runs the 8x8 DTW DP).
// ---------------------------------------------------------------------------
__global__ void __launch_bounds__(256) k_local(
    const float* __restrict__ localf,
    const int* __restrict__ p_inds, const int* __restrict__ n_inds,
    float* __restrict__ lap, float* __restrict__ lan) {
  const int lane = threadIdx.x & 31;
  const int wave = threadIdx.x >> 5;
  const int task = blockIdx.x * 8 + wave;       // exactly 2*N tasks
  const int i  = task >> 1;
  const int br = task & 1;
  const int j  = br ? n_inds[i] : p_inds[i];
  const float* X = localf + (size_t)i * (C_ * P_);
  const float* Y = localf + (size_t)j * (C_ * P_);
  const int m  = lane >> 2;
  const int n0 = (lane & 3) * 2, n1 = n0 + 1;
  float dot0 = 0.f, dot1 = 0.f, xx = 0.f, yy0 = 0.f, yy1 = 0.f;
  for (int c = 0; c < C_; ++c) {
    float xc = X[c * P_ + m];
    float y0 = Y[c * P_ + n0];
    float y1 = Y[c * P_ + n1];
    dot0 = fmaf(xc, y0, dot0);
    dot1 = fmaf(xc, y1, dot1);
    xx  = fmaf(xc, xc, xx);
    yy0 = fmaf(y0, y0, yy0);
    yy1 = fmaf(y1, y1, yy1);
  }
  float dm0 = tanhf(0.5f * sqrtf(fmaxf(xx + yy0 - 2.f * dot0, EPS_)));
  float dm1 = tanhf(0.5f * sqrtf(fmaxf(xx + yy1 - 2.f * dot1, EPS_)));
  __shared__ float sdm[8][8][8];  // [wave][m][n]
  sdm[wave][m][n0] = dm0;
  sdm[wave][m][n1] = dm1;
  __syncthreads();
  if (lane == 0) {
    float row[8];
    row[0] = sdm[wave][0][0];
#pragma unroll
    for (int c2 = 1; c2 < 8; ++c2) row[c2] = row[c2 - 1] + sdm[wave][0][c2];
    for (int r = 1; r < 8; ++r) {
      row[0] += sdm[wave][r][0];
#pragma unroll
      for (int c2 = 1; c2 < 8; ++c2)
        row[c2] = fminf(row[c2], row[c2 - 1]) + sdm[wave][r][c2];
    }
    (br ? lan : lap)[i] = row[7];
  }
}

// ---------------------------------------------------------------------------
// Kernel 4: margin-ranking losses (mean of relu(margin - (an - ap))).
// ---------------------------------------------------------------------------
__global__ void k_loss(const float* __restrict__ ap, const float* __restrict__ an,
                       const float* __restrict__ lap, const float* __restrict__ lan,
                       float* __restrict__ out) {
  float g = 0.f, l = 0.f;
  for (int i = threadIdx.x; i < N_; i += blockDim.x) {
    g += fmaxf(MARGIN_ - (an[i] - ap[i]), 0.f);
    l += fmaxf(MARGIN_ - (lan[i] - lap[i]), 0.f);
  }
  __shared__ float rg[256], rl[256];
  rg[threadIdx.x] = g;
  rl[threadIdx.x] = l;
  __syncthreads();
  for (int s = 128; s > 0; s >>= 1) {
    if (threadIdx.x < s) {
      rg[threadIdx.x] += rg[threadIdx.x + s];
      rl[threadIdx.x] += rl[threadIdx.x + s];
    }
    __syncthreads();
  }
  if (threadIdx.x == 0) {
    out[0] = rg[0] / (float)N_;
    out[1] = rl[0] / (float)N_;
  }
}

// ---------------------------------------------------------------------------
extern "C" void kernel_launch(void* const* d_in, const int* in_sizes, int n_in,
                              void* d_out, int out_size, void* d_ws, size_t ws_size,
                              hipStream_t stream) {
  const float* inputs  = (const float*)d_in[0];
  const int*   targets = (const int*)d_in[1];
  const float* localf  = (const float*)d_in[2];
  float* out = (float*)d_out;

  char* ws = (char*)d_ws;
  __bf16* featsb = (__bf16*)ws;
  size_t off = (size_t)N_ * D_ * sizeof(__bf16);          // 16 MB
  float* pAP = (float*)(ws + off); off += (size_t)SPLITS_ * N_ * sizeof(float);
  float* pAN = (float*)(ws + off); off += (size_t)SPLITS_ * N_ * sizeof(float);
  int*   pPI = (int*)(ws + off);   off += (size_t)SPLITS_ * N_ * sizeof(int);
  int*   pNI = (int*)(ws + off);   off += (size_t)SPLITS_ * N_ * sizeof(int);
  float* dist_ap = (float*)(ws + off); off += N_ * sizeof(float);
  float* dist_an = (float*)(ws + off); off += N_ * sizeof(float);
  int*   p_inds  = (int*)(ws + off);   off += N_ * sizeof(int);
  int*   n_inds  = (int*)(ws + off);   off += N_ * sizeof(int);
  float* lap     = (float*)(ws + off); off += N_ * sizeof(float);
  float* lan     = (float*)(ws + off); off += N_ * sizeof(float);

  k_normalize<<<dim3(N_), dim3(256), 0, stream>>>(inputs, featsb);
  k_gram_mine<<<dim3((N_ / BM_) * SPLITS_), dim3(256), 0, stream>>>(
      featsb, targets, pAP, pAN, pPI, pNI);
  k_mine_combine<<<dim3(N_ / 256), dim3(256), 0, stream>>>(
      pAP, pAN, pPI, pNI, dist_ap, dist_an, p_inds, n_inds);
  k_local<<<dim3(2 * N_ / 8), dim3(256), 0, stream>>>(localf, p_inds, n_inds, lap, lan);
  k_loss<<<dim3(1), dim3(256), 0, stream>>>(dist_ap, dist_an, lap, lan, out);
}
